// API_52613349376348
// MI455X (gfx1250) — matrix operand order repaired and checked by reading, not compiled, not run on hardware
//
#include <hip/hip_runtime.h>
#include <hip/hip_bf16.h>

typedef __attribute__((ext_vector_type(16))) _Float16 v16h;
typedef __attribute__((ext_vector_type(8)))  float    v8f;

#define NATOM  4096
#define EMBED  64
#define NRBF   16
#define MODEL  512
#define INFEAT 1088  // EMBED + NRBF*EMBED

// ---------------------------------------------------------------------------
// Kernel 1: center positions (single block).  pc[i] = pos[i] - mean(pos)
// ---------------------------------------------------------------------------
__global__ __launch_bounds__(256) void k_center(const float* __restrict__ pos,
                                                float* __restrict__ pc) {
  __shared__ float rx[256], ry[256], rz[256];
  const int t = threadIdx.x;
  float sx = 0.f, sy = 0.f, sz = 0.f;
  for (int i = t; i < NATOM; i += 256) {
    sx += pos[i * 3 + 0];
    sy += pos[i * 3 + 1];
    sz += pos[i * 3 + 2];
  }
  rx[t] = sx; ry[t] = sy; rz[t] = sz;
  __syncthreads();
  for (int off = 128; off > 0; off >>= 1) {
    if (t < off) { rx[t] += rx[t + off]; ry[t] += ry[t + off]; rz[t] += rz[t + off]; }
    __syncthreads();
  }
  const float mx = rx[0] * (1.0f / NATOM);
  const float my = ry[0] * (1.0f / NATOM);
  const float mz = rz[0] * (1.0f / NATOM);
  for (int i = t; i < NATOM; i += 256) {
    pc[i * 3 + 0] = pos[i * 3 + 0] - mx;
    pc[i * 3 + 1] = pos[i * 3 + 1] - my;
    pc[i * 3 + 2] = pos[i * 3 + 2] - mz;
  }
}

// ---------------------------------------------------------------------------
// Kernel 2: gather embeddings -> d-major f16 matrix E_T[64][4096] (WMMA B
// operand feed) + per-block partial column sums of the *f16-rounded* E
// (used both as the first 64 features and as the exact diagonal correction).
// ---------------------------------------------------------------------------
__global__ __launch_bounds__(256) void k_gather(const int* __restrict__ an,
                                                const float* __restrict__ etab,
                                                _Float16* __restrict__ eT,
                                                float* __restrict__ ecol_part) {
  const int t = threadIdx.x;
  const int d = t & 63;
  const int s = t >> 6;
  const int jbase = blockIdx.x * 256 + s * 64;
  float part = 0.f;
  for (int q = 0; q < 64; ++q) {
    const int j = jbase + q;
    const int a = an[j];
    const _Float16 hv = (_Float16)etab[a * EMBED + d];
    eT[(size_t)d * NATOM + j] = hv;
    part += (float)hv;
  }
  __shared__ float red[256];
  red[t] = part;
  __syncthreads();
  if (s == 0)
    ecol_part[blockIdx.x * 64 + d] = red[d] + red[64 + d] + red[128 + d] + red[192 + d];
}

// ---------------------------------------------------------------------------
// Kernel 3: the 34-GFLOP einsum on WMMA, 2x-unrolled ping-pong pipeline.
// Block = 16-row i-tile; wave w owns k = {2w, 2w+1} x 4 d-tiles.
// NO diagonal masking here: rbf(i,i)=1 contributes exactly sum_i E[i,d] to
// every k after the axis-0 reduction, subtracted once in k_reduce.
// ---------------------------------------------------------------------------
__global__ __launch_bounds__(256) void k_neighbor(const float* __restrict__ pc,
                                                  const _Float16* __restrict__ eT,
                                                  const float* __restrict__ gamma,
                                                  float* __restrict__ nbr_part) {
  __shared__ float spc[NATOM * 3];  // 48 KB of the 320 KB WGP LDS
  const int tid = threadIdx.x;
  for (int idx = tid; idx < NATOM * 3; idx += 256) spc[idx] = pc[idx];
  __syncthreads();

  const int wave = tid >> 5;       // 0..7
  const int lane = tid & 31;
  const int g    = lane >> 4;      // 16-lane group
  const int r    = lane & 15;      // row within tile (A: M, B/C: N)
  const int irow = blockIdx.x * 16 + r;

  const float pix = spc[irow * 3 + 0];
  const float piy = spc[irow * 3 + 1];
  const float piz = spc[irow * 3 + 2];

  const int   k0 = wave * 2;
  const float g0 = gamma[k0];
  const float g1 = gamma[k0 + 1];

  v8f acc[2][4] = {};  // [k-slot][d-tile], f32 16x16 accumulators

  // A-tile builder: 16-bit A (16x32) in contiguous k order, same as B: lane row M = lane&15,
  // half h -> K = g*16 + h.  Same-address LDS broadcasts
  // within each 16-lane group.
  auto buildA = [&](int j0, v16h& A0, v16h& A1) {
#pragma unroll
    for (int h = 0; h < 16; ++h) {
      const int K = g * 16 + h;
      const int j = j0 + K;
      const float dx = pix - spc[j * 3 + 0];
      const float dy = piy - spc[j * 3 + 1];
      const float dz = piz - spc[j * 3 + 2];
      const float d2 = dx * dx + dy * dy + dz * dz;
      A0[h] = (_Float16)__expf(-g0 * d2);
      A1[h] = (_Float16)__expf(-g1 * d2);
    }
  };

  // B layout (32x16 f16): lane col N = lane&15, half h -> K = g*16 + h
  // => one aligned 32-byte contiguous load per lane from d-major E_T.
  const _Float16* ebase = eT + (size_t)r * NATOM + g * 16;
  auto loadB4 = [&](int j0, v16h& X0, v16h& X1, v16h& X2, v16h& X3) {
    X0 = *(const v16h*)(ebase + j0);
    X1 = *(const v16h*)(ebase + (size_t)16 * NATOM + j0);
    X2 = *(const v16h*)(ebase + (size_t)32 * NATOM + j0);
    X3 = *(const v16h*)(ebase + (size_t)48 * NATOM + j0);
  };

  auto step = [&](int j0, const v16h& B0, const v16h& B1,
                          const v16h& B2, const v16h& B3) {
    v16h A0, A1;
    buildA(j0, A0, A1);
    acc[0][0] = __builtin_amdgcn_wmma_f32_16x16x32_f16(false, A0, false, B0, (short)0, acc[0][0], false, false);
    acc[1][0] = __builtin_amdgcn_wmma_f32_16x16x32_f16(false, A1, false, B0, (short)0, acc[1][0], false, false);
    acc[0][1] = __builtin_amdgcn_wmma_f32_16x16x32_f16(false, A0, false, B1, (short)0, acc[0][1], false, false);
    acc[1][1] = __builtin_amdgcn_wmma_f32_16x16x32_f16(false, A1, false, B1, (short)0, acc[1][1], false, false);
    acc[0][2] = __builtin_amdgcn_wmma_f32_16x16x32_f16(false, A0, false, B2, (short)0, acc[0][2], false, false);
    acc[1][2] = __builtin_amdgcn_wmma_f32_16x16x32_f16(false, A1, false, B2, (short)0, acc[1][2], false, false);
    acc[0][3] = __builtin_amdgcn_wmma_f32_16x16x32_f16(false, A0, false, B3, (short)0, acc[0][3], false, false);
    acc[1][3] = __builtin_amdgcn_wmma_f32_16x16x32_f16(false, A1, false, B3, (short)0, acc[1][3], false, false);
  };

  v16h Ba0, Ba1, Ba2, Ba3, Bb0, Bb1, Bb2, Bb3;
  loadB4(0, Ba0, Ba1, Ba2, Ba3);

  for (int j0 = 0; j0 < NATOM - 64; j0 += 64) {
    loadB4(j0 + 32, Bb0, Bb1, Bb2, Bb3);       // in flight during step(j0)
    __builtin_prefetch(ebase + j0 + 96, 0, 3); // warm the chunk after next
    step(j0, Ba0, Ba1, Ba2, Ba3);
    loadB4(j0 + 64, Ba0, Ba1, Ba2, Ba3);       // in flight during step(j0+32)
    step(j0 + 32, Bb0, Bb1, Bb2, Bb3);
  }
  // epilogue: last two chunks
  loadB4(NATOM - 32, Bb0, Bb1, Bb2, Bb3);
  step(NATOM - 64, Ba0, Ba1, Ba2, Ba3);
  step(NATOM - 32, Bb0, Bb1, Bb2, Bb3);

  // ---- reduce each C tile over M (the 16 i-rows of this block) ----------
  // C layout: lane holds col N = lane&15, VGPR rr -> row M = rr + 8*g.
#pragma unroll
  for (int kk = 0; kk < 2; ++kk) {
#pragma unroll
    for (int tt = 0; tt < 4; ++tt) {
      float s = 0.f;
#pragma unroll
      for (int rr = 0; rr < 8; ++rr) s += acc[kk][tt][rr];
      s += __shfl_xor(s, 16, 32);  // combine the two M-halves (wave32)
      if (lane < 16)
        nbr_part[(size_t)blockIdx.x * (NRBF * EMBED) + (k0 + kk) * EMBED + tt * 16 + lane] = s;
    }
  }
}

// ---------------------------------------------------------------------------
// Kernel 4: deterministic reduction of partials into colsum[1088].
// For the neighbor features, subtract ecol[d] once: that removes the
// rbf(i,i)=1 diagonal contribution (= sum_i E[i,d]) from every k.
// ---------------------------------------------------------------------------
__global__ __launch_bounds__(256) void k_reduce(const float* __restrict__ ecol_part,
                                                const float* __restrict__ nbr_part,
                                                float* __restrict__ colsum) {
  const int f = blockIdx.x * 256 + threadIdx.x;
  if (f >= INFEAT) return;
  float s = 0.f;
  if (f < EMBED) {
    for (int bk = 0; bk < 16; ++bk) s += ecol_part[bk * 64 + f];
  } else {
    const int q = f - EMBED;
    const int d = q & 63;
    float e = 0.f;
    for (int bk = 0; bk < 16; ++bk) e += ecol_part[bk * 64 + d];
    for (int bk = 0; bk < 256; ++bk) s += nbr_part[(size_t)bk * (NRBF * EMBED) + q];
    s -= e;  // diagonal correction
  }
  colsum[f] = s;
}

// ---------------------------------------------------------------------------
// Kernel 5: out = colsum @ W + NATOM * b   (1088x512 matvec, fp32)
// ---------------------------------------------------------------------------
__global__ __launch_bounds__(256) void k_final(const float* __restrict__ colsum,
                                               const float* __restrict__ W,
                                               const float* __restrict__ b,
                                               float* __restrict__ out) {
  const int m = blockIdx.x * 256 + threadIdx.x;
  if (m >= MODEL) return;
  float acc = (float)NATOM * b[m];
  for (int f = 0; f < INFEAT; ++f) acc += colsum[f] * W[(size_t)f * MODEL + m];
  out[m] = acc;
}

// ---------------------------------------------------------------------------
extern "C" void kernel_launch(void* const* d_in, const int* in_sizes, int n_in,
                              void* d_out, int out_size, void* d_ws, size_t ws_size,
                              hipStream_t stream) {
  const int*   an    = (const int*)d_in[0];    // [4096]
  const float* pos   = (const float*)d_in[1];  // [4096,3]
  const float* etab  = (const float*)d_in[2];  // [128,64]
  const float* gamma = (const float*)d_in[3];  // [16]
  const float* W     = (const float*)d_in[4];  // [1088,512]
  const float* b     = (const float*)d_in[5];  // [512]
  float*       out   = (float*)d_out;          // [512]

  char* ws = (char*)d_ws;
  float*    pc        = (float*)(ws);                            //  48 KB
  _Float16* eT        = (_Float16*)(ws + 64 * 1024);             // 512 KB
  float*    nbr_part  = (float*)(ws + 576 * 1024);               // 256*1024*4 = 1 MB
  float*    ecol_part = (float*)(ws + 1600 * 1024);              // 16*64*4
  float*    colsum    = (float*)(ws + 1608 * 1024);              // 1088*4

  hipLaunchKernelGGL(k_center,   dim3(1),   dim3(256), 0, stream, pos, pc);
  hipLaunchKernelGGL(k_gather,   dim3(16),  dim3(256), 0, stream, an, etab, eT, ecol_part);
  hipLaunchKernelGGL(k_neighbor, dim3(256), dim3(256), 0, stream, pc, eT, gamma, nbr_part);
  hipLaunchKernelGGL(k_reduce,   dim3(5),   dim3(256), 0, stream, ecol_part, nbr_part, colsum);
  hipLaunchKernelGGL(k_final,    dim3(2),   dim3(256), 0, stream, colsum, W, b, out);
}
